// ExpertRouter_79388175499573
// MI455X (gfx1250) — matrix operand order repaired and checked by reading, not compiled, not run on hardware
//
#include <hip/hip_runtime.h>
#include <hip/hip_bf16.h>

typedef __attribute__((ext_vector_type(16))) __bf16 v16bf;
typedef __attribute__((ext_vector_type(8)))  __bf16 v8bf;
typedef __attribute__((ext_vector_type(8)))  float  v8f;

#define BATCH 16384
#define DIM   4096
#define HID   2048
#define NEXP  64

static __device__ inline v16bf make_v16(v8bf lo, v8bf hi) {
  v16bf r;
#pragma unroll
  for (int i = 0; i < 8; ++i) { r[i] = lo[i]; r[i + 8] = hi[i]; }
  return r;
}

// ---------------------------------------------------------------------------
// One-time converts: x -> bf16, W1 -> bf16 transposed [HID][DIM],
// W2 -> bf16 transposed [NEXP][HID].
// ---------------------------------------------------------------------------
__global__ __launch_bounds__(256) void convert_x_kernel(
    const float* __restrict__ x, __bf16* __restrict__ xb)
{
  const size_t i = ((size_t)blockIdx.x * 256 + threadIdx.x) * 8;
  const float4 a = *(const float4*)&x[i];
  const float4 b = *(const float4*)&x[i + 4];
  v8bf o;
  o[0] = (__bf16)a.x; o[1] = (__bf16)a.y; o[2] = (__bf16)a.z; o[3] = (__bf16)a.w;
  o[4] = (__bf16)b.x; o[5] = (__bf16)b.y; o[6] = (__bf16)b.z; o[7] = (__bf16)b.w;
  *(v8bf*)&xb[i] = o;
}

__global__ __launch_bounds__(256) void convert_w1t_kernel(
    const float* __restrict__ W1, __bf16* __restrict__ W1T)
{
  __shared__ __bf16 tile[64][64];
  const int kb  = blockIdx.x * 64;   // k tile base (DIM)
  const int nb0 = blockIdx.y * 64;   // n tile base (HID)
  const int t   = threadIdx.x;
  {
    const int kk = t >> 2;
    const int nn = (t & 3) * 16;
    const float* src = &W1[(size_t)(kb + kk) * HID + nb0 + nn];
#pragma unroll
    for (int j = 0; j < 16; ++j) tile[kk][nn + j] = (__bf16)src[j];
  }
  __syncthreads();
  {
    const int nn2 = t >> 2;
    const int kk2 = (t & 3) * 16;
    v8bf o0, o1;
#pragma unroll
    for (int j = 0; j < 8; ++j) {
      o0[j] = tile[kk2 + j][nn2];
      o1[j] = tile[kk2 + 8 + j][nn2];
    }
    __bf16* dst = &W1T[(size_t)(nb0 + nn2) * DIM + kb + kk2];
    *(v8bf*)dst       = o0;
    *(v8bf*)(dst + 8) = o1;
  }
}

__global__ __launch_bounds__(256) void convert_w2t_kernel(
    const float* __restrict__ W2, __bf16* __restrict__ W2T)
{
  const int n = blockIdx.x;
  for (int k = threadIdx.x; k < HID; k += 256)
    W2T[(size_t)n * HID + k] = (__bf16)W2[(size_t)k * NEXP + n];
}

// ---------------------------------------------------------------------------
// Kernel A: h = silu(xb @ W1T^T + b1), bf16 out [BATCH][HID]
// 256 threads (8 waves); block tile 128x128; wave tile 64x32 (4x2 WMMA tiles).
// Double-buffered LDS, single loop body (unroll disabled) so the 8
// accumulators stay pinned in their WMMA tuples (no acc copies / hazard NOPs).
// ---------------------------------------------------------------------------
#define LDSK 40  // padded k stride (elements) to spread LDS banks

__global__ __launch_bounds__(256) void gemm1_silu_kernel(
    const __bf16* __restrict__ xb, const __bf16* __restrict__ W1T,
    const float* __restrict__ b1, __bf16* __restrict__ h)
{
  __shared__ __align__(16) __bf16 xs[2][128][LDSK];   // [buf][row][k]
  __shared__ __align__(16) __bf16 wsd[2][128][LDSK];  // [buf][n][k]

  const int t    = threadIdx.x;
  const int wave = t >> 5;
  const int lane = t & 31;
  const int mb   = blockIdx.y * 128;
  const int nb   = blockIdx.x * 128;
  const int wave_m = (wave >> 2) * 64;   // 0 / 64
  const int wave_n = (wave & 3) * 32;    // 0 / 32 / 64 / 96
  const int  lr  = lane & 15;
  const bool hiL = lane >= 16;

  // staging: 256 threads copy 128 rows x 32 bf16 (32B per thread per array)
  const int sr = t >> 1;
  const int sc = (t & 1) * 16;
  const __bf16* gx = &xb[(size_t)(mb + sr) * DIM + sc];
  const __bf16* gw = &W1T[(size_t)(nb + sr) * DIM + sc];

  v8f acc[4][2];
#pragma unroll
  for (int i = 0; i < 4; ++i)
#pragma unroll
    for (int j = 0; j < 2; ++j) acc[i][j] = (v8f){};

  // prologue: stage chunk 0 into buffer 0
  *(uint4*)&xs[0][sr][sc]      = *(const uint4*)(gx);
  *(uint4*)&xs[0][sr][sc + 8]  = *(const uint4*)(gx + 8);
  *(uint4*)&wsd[0][sr][sc]     = *(const uint4*)(gw);
  *(uint4*)&wsd[0][sr][sc + 8] = *(const uint4*)(gw + 8);
  __syncthreads();

  const int NITER = DIM / 32;
  int buf = 0;
#pragma clang loop unroll(disable)
  for (int it = 0; it < NITER; ++it) {
    const int obuf = buf ^ 1;
    if (it + 1 < NITER) {  // stage next chunk while computing current
      const __bf16* px = gx + (size_t)(it + 1) * 32;
      const __bf16* pw = gw + (size_t)(it + 1) * 32;
      *(uint4*)&xs[obuf][sr][sc]      = *(const uint4*)(px);
      *(uint4*)&xs[obuf][sr][sc + 8]  = *(const uint4*)(px + 8);
      *(uint4*)&wsd[obuf][sr][sc]     = *(const uint4*)(pw);
      *(uint4*)&wsd[obuf][sr][sc + 8] = *(const uint4*)(pw + 8);
    }
    // B fragments (lane = column, K contiguous per half-wave)
    const __bf16* bp0 = &wsd[buf][wave_n + lr][hiL ? 16 : 0];
    const __bf16* bp1 = &wsd[buf][wave_n + 16 + lr][hiL ? 16 : 0];
    v16bf B0 = make_v16(*(const v8bf*)bp0, *(const v8bf*)(bp0 + 8));
    v16bf B1 = make_v16(*(const v8bf*)bp1, *(const v8bf*)(bp1 + 8));
#pragma unroll
    for (int i = 0; i < 4; ++i) {
      const int row = wave_m + i * 16 + lr;
      v8bf alo = *(const v8bf*)&xs[buf][row][hiL ? 16 : 0];
      v8bf ahi = *(const v8bf*)&xs[buf][row][hiL ? 24 : 8];
      v16bf A = make_v16(alo, ahi);
      acc[i][0] = __builtin_amdgcn_wmma_f32_16x16x32_bf16(
          false, A, false, B0, (short)0, acc[i][0], false, false);
      acc[i][1] = __builtin_amdgcn_wmma_f32_16x16x32_bf16(
          false, A, false, B1, (short)0, acc[i][1], false, false);
    }
    __syncthreads();
    buf = obuf;
  }

  // epilogue: bias + SiLU, store bf16 h
#pragma unroll
  for (int j = 0; j < 2; ++j) {
    const int   n    = nb + wave_n + j * 16 + lr;
    const float bias = b1[n];
#pragma unroll
    for (int i = 0; i < 4; ++i) {
      const int mrow = mb + wave_m + i * 16 + (hiL ? 8 : 0);
#pragma unroll
      for (int v = 0; v < 8; ++v) {
        float val = acc[i][j][v] + bias;
        float s   = val / (1.0f + __expf(-val));
        h[(size_t)(mrow + v) * HID + n] = (__bf16)s;
      }
    }
  }
}

// ---------------------------------------------------------------------------
// Kernel B: logits = h @ W2T^T + b2 via WMMA with fragments loaded directly
// from global (h rows + W2T are cache-resident; per-lane loads give A and B the same
// contiguous k = 16h + s order), then top-2 softmax gating + aux-loss stats.
// Block: 128 threads (4 waves), 16 rows; wave w covers experts w*16..+15.
// ---------------------------------------------------------------------------
__global__ __launch_bounds__(128) void gemm2_router_kernel(
    const __bf16* __restrict__ h, const __bf16* __restrict__ W2T,
    const float* __restrict__ b2, float* __restrict__ wout,
    float* __restrict__ iout, float* __restrict__ freq_g,
    float* __restrict__ psum_g)
{
  __shared__ float lds_log[16][NEXP];
  __shared__ float rmax[16], rinv[16];
  __shared__ float lds_freq[NEXP];

  const int t    = threadIdx.x;
  const int wave = t >> 5;
  const int lane = t & 31;
  const int rb   = blockIdx.x * 16;
  const int  lr  = lane & 15;
  const bool hiL = lane >= 16;
  const int  wn  = wave * 16;

  const __bf16* arow = &h[(size_t)(rb + lr) * HID];
  const __bf16* brow = &W2T[(size_t)(wn + lr) * HID];

  v8f acc = {};
  for (int k0 = 0; k0 < HID; k0 += 32) {
    v8bf alo = *(const v8bf*)(arow + k0 + (hiL ? 16 : 0));
    v8bf ahi = *(const v8bf*)(arow + k0 + (hiL ? 24 : 8));
    v16bf A = make_v16(alo, ahi);
    const __bf16* bp = brow + k0 + (hiL ? 16 : 0);
    v16bf Bm = make_v16(*(const v8bf*)bp, *(const v8bf*)(bp + 8));
    acc = __builtin_amdgcn_wmma_f32_16x16x32_bf16(
        false, A, false, Bm, (short)0, acc, false, false);
  }

  // scatter logits (+bias) into LDS [16 rows][64 experts]
  {
    const int   n    = wn + lr;
    const float bias = b2[n];
    const int   mb2  = hiL ? 8 : 0;
#pragma unroll
    for (int v = 0; v < 8; ++v) lds_log[mb2 + v][n] = acc[v] + bias;
  }
  if (t < NEXP) lds_freq[t] = 0.0f;
  __syncthreads();

  // per-row gating: top-2, softmax over top-2, full-softmax row stats
  if (t < 16) {
    const int row = rb + t;
    float v1 = -1e30f, v2 = -1e30f;
    int   i1 = 0, i2 = 0;
    for (int j = 0; j < NEXP; ++j) {
      float l = lds_log[t][j];
      if (l > v1)      { v2 = v1; i2 = i1; v1 = l; i1 = j; }
      else if (l > v2) { v2 = l; i2 = j; }
    }
    float s = 0.0f;
    for (int j = 0; j < NEXP; ++j) s += __expf(lds_log[t][j] - v1);
    rmax[t] = v1;
    rinv[t] = 1.0f / s;

    const float e2 = __expf(v2 - v1);
    const float wa = 1.0f / (1.0f + e2);
    wout[row * 2 + 0] = wa;
    wout[row * 2 + 1] = e2 * wa;
    iout[row * 2 + 0] = (float)i1;
    iout[row * 2 + 1] = (float)i2;
    atomicAdd(&lds_freq[i1], 1.0f);
  }
  __syncthreads();

  // per-expert column sums of softmax probs; flush block stats
  if (t < NEXP) {
    float s = 0.0f;
#pragma unroll
    for (int m = 0; m < 16; ++m)
      s += __expf(lds_log[m][t] - rmax[m]) * rinv[m];
    atomicAdd(&psum_g[t], s);
    const float f = lds_freq[t];
    if (f != 0.0f) atomicAdd(&freq_g[t], f);
  }
}

// ---------------------------------------------------------------------------
__global__ void zero_stats_kernel(float* __restrict__ s) {
  s[threadIdx.x] = 0.0f;  // 128 threads: freq[64] + psum[64]
}

__global__ void finalize_kernel(const float* __restrict__ freq_g,
                                const float* __restrict__ psum_g,
                                float* __restrict__ aux) {
  __shared__ float red[NEXP];
  const int t = threadIdx.x;
  red[t] = freq_g[t] * psum_g[t];
  __syncthreads();
  for (int s = 32; s > 0; s >>= 1) {
    if (t < s) red[t] += red[t + s];
    __syncthreads();
  }
  if (t == 0)
    aux[0] = (float)NEXP * red[0] / ((float)BATCH * (float)BATCH);
}

// ---------------------------------------------------------------------------
extern "C" void kernel_launch(void* const* d_in, const int* in_sizes, int n_in,
                              void* d_out, int out_size, void* d_ws, size_t ws_size,
                              hipStream_t stream) {
  const float* x  = (const float*)d_in[0];
  const float* W1 = (const float*)d_in[1];
  const float* b1 = (const float*)d_in[2];
  const float* W2 = (const float*)d_in[3];
  const float* b2 = (const float*)d_in[4];

  float* out  = (float*)d_out;
  float* wout = out;              // [B,2] gate weights
  float* iout = out + 2 * BATCH;  // [B,2] top-k indices (as float)
  float* aux  = out + 4 * BATCH;  // [1]   aux loss

  // workspace layout
  const size_t XB_BYTES  = (size_t)BATCH * DIM  * 2;  // 128 MB
  const size_t W1T_BYTES = (size_t)HID   * DIM  * 2;  //  16 MB
  const size_t W2T_BYTES = (size_t)NEXP  * HID  * 2;  // 256 KB
  const size_t H_BYTES   = (size_t)BATCH * HID  * 2;  //  64 MB
  char* ws = (char*)d_ws;
  __bf16* xb   = (__bf16*)ws;
  __bf16* W1T  = (__bf16*)(ws + XB_BYTES);
  __bf16* W2T  = (__bf16*)(ws + XB_BYTES + W1T_BYTES);
  __bf16* hbuf = (__bf16*)(ws + XB_BYTES + W1T_BYTES + W2T_BYTES);
  float*  stats  = (float*)(ws + XB_BYTES + W1T_BYTES + W2T_BYTES + H_BYTES);
  float*  freq_g = stats;
  float*  psum_g = stats + NEXP;

  // one-time precision/layout converts
  convert_x_kernel<<<(size_t)BATCH * DIM / 8 / 256, 256, 0, stream>>>(x, xb);
  dim3 gw1(DIM / 64, HID / 64);
  convert_w1t_kernel<<<gw1, 256, 0, stream>>>(W1, W1T);
  convert_w2t_kernel<<<NEXP, 256, 0, stream>>>(W2, W2T);

  zero_stats_kernel<<<1, 128, 0, stream>>>(stats);

  dim3 g1(HID / 128, BATCH / 128);  // (16, 128)
  gemm1_silu_kernel<<<g1, 256, 0, stream>>>(xb, W1T, b1, hbuf);

  gemm2_router_kernel<<<BATCH / 16, 128, 0, stream>>>(hbuf, W2T, b2, wout,
                                                      iout, freq_g, psum_g);

  finalize_kernel<<<1, NEXP, 0, stream>>>(freq_g, psum_g, aux);
}
